// DecoupledInfoNCELoss_50525995270621
// MI455X (gfx1250) — compile-verified, matrix operand order checked by reading, not run on hardware
//
#include <hip/hip_runtime.h>
#include <hip/hip_bf16.h>

typedef __attribute__((ext_vector_type(16))) _Float16 v16h;
typedef __attribute__((ext_vector_type(8)))  float    v8f;

#define N_ROWS   8192
#define DIM      256
#define NCHUNK   8                           // column chunks (grid.y)
#define COLS_PER_CHUNK (N_ROWS / NCHUNK)     // 1024
#define TILES_PER_CHUNK (COLS_PER_CHUNK / 16) // 64
// exp((dot-1)*10) computed as exp2((dot-1)*10*log2(e))
#define K_EXP2   14.4269504088896340736f     // 10 * log2(e)

union V16 { v16h v; uint4 q[2]; };

// ---------------------------------------------------------------------------
// Kernel 1: row-normalize fp32 embeddings -> f16, one wave32 per row
// ---------------------------------------------------------------------------
__global__ __launch_bounds__(256)
void normalize_f16_kernel(const float* __restrict__ emb, _Float16* __restrict__ nrm) {
  const int row  = (blockIdx.x * blockDim.x + threadIdx.x) >> 5;
  const int lane = threadIdx.x & 31;
  const float* rp = emb + (size_t)row * DIM + lane * 8;
  float4 f0 = *(const float4*)(rp);
  float4 f1 = *(const float4*)(rp + 4);
  float ss = f0.x*f0.x + f0.y*f0.y + f0.z*f0.z + f0.w*f0.w
           + f1.x*f1.x + f1.y*f1.y + f1.z*f1.z + f1.w*f1.w;
  #pragma unroll
  for (int off = 16; off > 0; off >>= 1) ss += __shfl_xor(ss, off, 32);
  const float s = 1.0f / fmaxf(sqrtf(ss), 1e-8f);
  union { _Float16 h[8]; uint4 q; } o;
  o.h[0] = (_Float16)(f0.x * s); o.h[1] = (_Float16)(f0.y * s);
  o.h[2] = (_Float16)(f0.z * s); o.h[3] = (_Float16)(f0.w * s);
  o.h[4] = (_Float16)(f1.x * s); o.h[5] = (_Float16)(f1.y * s);
  o.h[6] = (_Float16)(f1.z * s); o.h[7] = (_Float16)(f1.w * s);
  *(uint4*)(nrm + (size_t)row * DIM + lane * 8) = o.q;
}

// ---------------------------------------------------------------------------
// Async stage of one 8KB column tile (16 rows x 256 f16, contiguous) into LDS.
// Each of 256 threads issues two GLOBAL_LOAD_ASYNC_TO_LDS_B128; INST_OFFSET
// is added to both the global and LDS addresses, so offset:4096 covers the
// second half. Tracked by ASYNCcnt.
// ---------------------------------------------------------------------------
__device__ __forceinline__ void stage_async(const _Float16* gsrc, unsigned lds_off,
                                            int tid) {
  unsigned long long g = (unsigned long long)(uintptr_t)gsrc + (unsigned long long)tid * 16;
  unsigned l = lds_off + (unsigned)tid * 16;
  asm volatile("global_load_async_to_lds_b128 %0, %1, off\n\t"
               "global_load_async_to_lds_b128 %0, %1, off offset:4096"
               :: "v"(l), "v"(g) : "memory");
}

__device__ __forceinline__ void wait_async_all() {
  asm volatile("s_wait_asynccnt 0x0" ::: "memory");
}

// ---------------------------------------------------------------------------
// Kernel 2: WMMA similarity + fused masked exp-accumulate.
// grid.x = 64 row-groups (128 rows, 8 waves x 16 rows),
// grid.y = 8 column chunks of 1024 columns.
// Double-buffered async LDS staging; one barrier per tile.
// ---------------------------------------------------------------------------
__global__ __launch_bounds__(256)
void infonce_wmma_kernel(const _Float16* __restrict__ nrm,
                         const int* __restrict__ target,
                         float* __restrict__ part_neg,
                         float* __restrict__ part_pos) {
  __shared__ _Float16 sB[2][16 * DIM];     // 2 x 8 KB double buffer
  __shared__ int      sTc[COLS_PER_CHUNK]; // 4 KB chunk targets

  const int tid  = threadIdx.x;
  const int lane = tid & 31;
  const int wave = tid >> 5;
  const int half = lane >> 4;              // 0: lanes 0-15, 1: lanes 16-31
  const int l16  = lane & 15;
  const int kb   = half * 8;               // K sub-offset per ISA A/B layout

  // LDS byte offset of sB (low 32 bits of shared-aperture address)
  const unsigned sb_off = (unsigned)(uintptr_t)(void*)&sB[0][0];

  const int jbase = blockIdx.y * COLS_PER_CHUNK;
  for (int i = tid; i < COLS_PER_CHUNK; i += 256) sTc[i] = target[jbase + i];

  // This wave's 16 rows; full-K A fragments (8 k-steps, 64 VGPRs)
  const int i0 = (blockIdx.x * 8 + wave) * 16;
  const _Float16* arow = nrm + (size_t)(i0 + l16) * DIM;
  V16 a[8];
  #pragma unroll
  for (int k = 0; k < 8; ++k) {
    a[k].q[0] = *(const uint4*)(arow + 32 * k + kb);
    a[k].q[1] = *(const uint4*)(arow + 32 * k + kb + 16);
  }

  const int rowbase = i0 + half * 8;       // C layout: reg r -> row rowbase+r
  int ti[8];
  #pragma unroll
  for (int r = 0; r < 8; ++r) ti[r] = target[rowbase + r];

  float s_neg[8], s_pos[8];
  #pragma unroll
  for (int r = 0; r < 8; ++r) { s_neg[r] = 0.0f; s_pos[r] = 0.0f; }

  // prologue: stage tile 0 into buffer 0
  stage_async(nrm + (size_t)jbase * DIM, sb_off, tid);
  wait_async_all();
  __syncthreads();

  for (int jt = 0; jt < TILES_PER_CHUNK; ++jt) {
    // stage next tile into the other buffer while computing this one
    if (jt + 1 < TILES_PER_CHUNK)
      stage_async(nrm + (size_t)(jbase + (jt + 1) * 16) * DIM,
                  sb_off + ((unsigned)((jt + 1) & 1)) * (16 * DIM * 2), tid);

    // preload all 8 B fragments, then 8 back-to-back WMMAs
    const _Float16* brow = &sB[jt & 1][0] + l16 * DIM;
    V16 b[8];
    #pragma unroll
    for (int k = 0; k < 8; ++k) {
      b[k].q[0] = *(const uint4*)(brow + 32 * k + kb);
      b[k].q[1] = *(const uint4*)(brow + 32 * k + kb + 16);
    }
    v8f c = {};
    #pragma unroll
    for (int k = 0; k < 8; ++k)
      c = __builtin_amdgcn_wmma_f32_16x16x32_f16(false, a[k].v, false, b[k].v,
                                                 (short)0, c, false, false);

    // fused masked accumulation with fixed LSE max (sim/T <= 10 always)
    const int j0g  = jbase + jt * 16;
    const int tj   = sTc[jt * 16 + l16];
    const int colg = j0g + l16;
    #pragma unroll
    for (int r = 0; r < 8; ++r) {
      const float e   = __builtin_amdgcn_exp2f(__builtin_fmaf(c[r], K_EXP2, -K_EXP2));
      const bool  pos = (tj == ti[r]);
      const bool  dg  = (colg == rowbase + r);
      s_neg[r] += pos ? 0.0f : e;
      s_pos[r] += (pos && !dg) ? e : 0.0f;
    }

    wait_async_all();
    __syncthreads();
  }

  // reduce each row across its 16 lanes (butterfly), write chunk partials
  const size_t pb = (size_t)blockIdx.y * N_ROWS;
  #pragma unroll
  for (int r = 0; r < 8; ++r) {
    float sn = s_neg[r], sp = s_pos[r];
    #pragma unroll
    for (int off = 1; off < 16; off <<= 1) {
      sn += __shfl_xor(sn, off, 32);
      sp += __shfl_xor(sp, off, 32);
    }
    if (l16 == 0) {
      part_neg[pb + rowbase + r] = sn;
      part_pos[pb + rowbase + r] = sp;
    }
  }
}

// ---------------------------------------------------------------------------
// Kernel 3: deterministic final reduction -> scalar loss
// loss = sum_i [ log(sum_neg_i) - log(sum_pos_i) ]   (fixed-max offsets cancel)
// ---------------------------------------------------------------------------
__global__ __launch_bounds__(256)
void finalize_kernel(const float* __restrict__ part_neg,
                     const float* __restrict__ part_pos,
                     float* __restrict__ out) {
  __shared__ float red[256];
  const int tid = threadIdx.x;
  float acc = 0.0f;
  for (int row = tid; row < N_ROWS; row += 256) {
    float sn = 0.0f, sp = 0.0f;
    #pragma unroll
    for (int c = 0; c < NCHUNK; ++c) {
      sn += part_neg[(size_t)c * N_ROWS + row];
      sp += part_pos[(size_t)c * N_ROWS + row];
    }
    acc += __logf(sn) - __logf(sp);
  }
  red[tid] = acc;
  __syncthreads();
  #pragma unroll
  for (int s = 128; s > 0; s >>= 1) {
    if (tid < s) red[tid] += red[tid + s];
    __syncthreads();
  }
  if (tid == 0) out[0] = red[0];
}

// ---------------------------------------------------------------------------
extern "C" void kernel_launch(void* const* d_in, const int* in_sizes, int n_in,
                              void* d_out, int out_size, void* d_ws, size_t ws_size,
                              hipStream_t stream) {
  const float* emb    = (const float*)d_in[0];
  const int*   target = (const int*)d_in[1];
  float*       out    = (float*)d_out;

  // workspace layout: [4 MB f16 normalized matrix][NCHUNK*N neg][NCHUNK*N pos]
  _Float16* nrm = (_Float16*)d_ws;
  float* part_neg = (float*)((char*)d_ws + (size_t)N_ROWS * DIM * sizeof(_Float16));
  float* part_pos = part_neg + (size_t)NCHUNK * N_ROWS;

  normalize_f16_kernel<<<N_ROWS / 8, 256, 0, stream>>>(emb, nrm);

  dim3 grid(N_ROWS / 128, NCHUNK);   // 64 row-groups x 8 column chunks
  infonce_wmma_kernel<<<grid, 256, 0, stream>>>(nrm, target, part_neg, part_pos);

  finalize_kernel<<<1, 256, 0, stream>>>(part_neg, part_pos, out);
}
